// LinearMultiheadAttention_40630390620257
// MI455X (gfx1250) — compile-verified, matrix operand order checked by reading, not run on hardware
//
#include <hip/hip_runtime.h>

// ---------------------------------------------------------------------------
// Problem constants (from reference):
//   B=2, S=2048, DM=1024, H=16, F=16, DH=64
//   feature dim D = 1 + F + F*F = 273, padded to DP = 288 (18 * 16)
//   chunk length CH = 64, extended value width NE = 80 (64 V cols + ones col)
// ---------------------------------------------------------------------------
constexpr int Bc  = 2;
constexpr int Sc  = 2048;
constexpr int DMc = 1024;
constexpr int Hc  = 16;
constexpr int Fc  = 16;
constexpr int DHc = 64;
constexpr int CHc = 64;
constexpr int DPc = 288;
constexpr int NEc = 80;

typedef __attribute__((ext_vector_type(16))) _Float16 v16h;
typedef __attribute__((ext_vector_type(8)))  _Float16 v8h;
typedef __attribute__((ext_vector_type(8)))  float    v8f;

#define DEV static __device__ __forceinline__

DEV v8f wmma_f32_f16(v16h a, v16h b, v8f c) {
  // D = A(16x32,f16) * B(32x16,f16) + C(16x16,f32)
  return __builtin_amdgcn_wmma_f32_16x16x32_f16(
      /*neg_a=*/false, a, /*neg_b=*/false, b,
      /*c_mod=*/(short)0, c, /*reuse_a=*/false, /*reuse_b=*/false);
}

// ---------------------------------------------------------------------------
// Vectorized fragment load. Works for:
//   * A fragment (16x32 MxK) from a row-major [M,K] source  (major = M)
//   * B fragment (32x16 KxN) from a TRANSPOSED [N,K] source (major = N)
// Per-lane data is contiguous in K: halves kb..kb+7 and kb+16..kb+23,
// so each fragment is exactly two 16-byte loads + one shuffle.
// Requires ld % 8 == 0 and 16-byte aligned base (all our operands satisfy).
// ---------------------------------------------------------------------------
DEV v16h load_frag_vec(const _Float16* src, int major0, int k0, int ld, int lane) {
  const int mj = major0 + (lane & 15);
  const int kb = k0 + ((lane >> 4) << 3);   // upper half-wave: K base += 8
  const _Float16* p = src + mj * ld + kb;
  const v8h lo = *(const v8h*)(p);          // K = kb .. kb+7
  const v8h hi = *(const v8h*)(p + 16);     // K = kb+16 .. kb+23
  return __builtin_shufflevector(lo, hi, 0, 1, 2, 3, 4, 5, 6, 7,
                                 8, 9, 10, 11, 12, 13, 14, 15);
}

// ---------------------------------------------------------------------------
// f32 -> f16 conversion (grid-stride), same layout
// ---------------------------------------------------------------------------
__global__ void cvt_f32_to_f16_kernel(const float* __restrict__ src,
                                      _Float16* __restrict__ dst, int n) {
  for (int i = blockIdx.x * blockDim.x + threadIdx.x; i < n;
       i += gridDim.x * blockDim.x)
    dst[i] = (_Float16)src[i];
}

// f32 [K,N] -> f16 transposed [N,K]  (weights: y = x @ W  ->  B^T layout)
__global__ void cvt_transpose_f16_kernel(const float* __restrict__ src,
                                         _Float16* __restrict__ dst,
                                         int K, int N) {
  const int total = K * N;
  for (int i = blockIdx.x * blockDim.x + threadIdx.x; i < total;
       i += gridDim.x * blockDim.x) {
    const int n = i / K;
    const int k = i - n * K;
    dst[i] = (_Float16)src[k * N + n];
  }
}

// ---------------------------------------------------------------------------
// WMMA GEMM: C[M,N](f32) = A[M,K](f16 row-major) * B(f16, given as B^T [N,K])
// 256 threads = 8 waves; each wave owns one 16x64 output patch.
// M % 16 == 0, N % 64 == 0, K % 32 == 0.
// ---------------------------------------------------------------------------
__global__ __launch_bounds__(256)
void wmma_gemm_kernel(const _Float16* __restrict__ A, const _Float16* __restrict__ BT,
                      float* __restrict__ C, int M, int N, int K) {
  const int wave = threadIdx.x >> 5;
  const int lane = threadIdx.x & 31;
  const int nP   = N >> 6;                    // 64-wide patches per row
  const int p    = blockIdx.x * 8 + wave;
  const int pr   = p / nP;
  const int pc   = p - pr * nP;
  if (pr >= (M >> 4)) return;
  const int r0 = pr << 4;
  const int c0 = pc << 6;

  v8f acc[4] = {{}, {}, {}, {}};
  for (int k = 0; k < K; k += 32) {
    if (k + 32 < K) {  // speculative prefetch -> global_prefetch_b8
      __builtin_prefetch(&A[(r0 + (lane & 15)) * K + k + 32], 0, 1);
      __builtin_prefetch(&BT[(c0 + (lane & 15)) * K + k + 32], 0, 1);
    }
    const v16h a = load_frag_vec(A, r0, k, K, lane);
#pragma unroll
    for (int u = 0; u < 4; ++u) {
      const v16h b = load_frag_vec(BT, c0 + u * 16, k, K, lane);
      acc[u] = wmma_f32_f16(a, b, acc[u]);
    }
  }
  const int n  = lane & 15;
  const int hi = (lane >> 4) << 3;            // upper half-wave: rows 8..15
#pragma unroll
  for (int u = 0; u < 4; ++u)
#pragma unroll
    for (int r = 0; r < 8; ++r)
      C[(r0 + r + hi) * N + c0 + u * 16 + n] = acc[u][r];
}

// ---------------------------------------------------------------------------
// Chunked causal linear attention, one workgroup per (b, h).
// 512 threads = 16 waves. State recurrence over 32 chunks of 64 rows.
//
// Per chunk:
//   phi_q [64x288], phi_k [64x288], phi_kT [288x64]   (Taylor map, padded)
//   A          = tril(phi_q @ phi_k^T)                (GEMM1 -> f16 LDS)
//   O_ext      = A @ V_ext + phi_q @ State_ext        (GEMM2, fused)
//                cols 0..63 numerator, col 64 = causal denominator
//   State_ext += phi_k^T @ V_ext                      (GEMM3, f32 regs)
// V_ext = [V | 1 | 0...] kept transposed [80x64]; state mirror transposed
// [80x288] so every WMMA operand loads as two B128s per lane.
// ---------------------------------------------------------------------------
__global__ __launch_bounds__(512)
void linattn_kernel(const float* __restrict__ Qg, const float* __restrict__ Kg,
                    const float* __restrict__ Vg, _Float16* __restrict__ Og) {
  const int bh = blockIdx.x;
  const int b  = bh >> 4;
  const int h  = bh & 15;

  extern __shared__ char smem[];
  float*    sm_q   = (float*)smem;                 // [64][16]
  float*    sm_k   = sm_q + CHc * Fc;              // [64][16]
  float*    sm_den = sm_k + CHc * Fc;              // [64]
  _Float16* phi_q  = (_Float16*)(sm_den + CHc);    // [64][288]
  _Float16* phi_k  = phi_q + CHc * DPc;            // [64][288]
  _Float16* phi_kT = phi_k + CHc * DPc;            // [288][64]
  _Float16* v_extT = phi_kT + DPc * CHc;           // [80][64]
  _Float16* a_m    = v_extT + NEc * CHc;           // [64][64]
  _Float16* st16T  = a_m + CHc * CHc;              // [80][288]

  const int tid  = threadIdx.x;
  const int wave = tid >> 5;
  const int lane = tid & 31;

  // zero transposed state mirror in LDS
  for (int i = tid; i < NEc * DPc; i += 512) st16T[i] = (_Float16)0.0f;

  // per-wave register state: wave w owns state tiles {w + 16*j}, 90 tiles total
  v8f stacc[6];
#pragma unroll
  for (int j = 0; j < 6; ++j) stacc[j] = (v8f){};

  const float inv_xn  = 0.5f;                   // 1/sqrt(sqrt(16))
  const float inv_x2n = 0.17677669529663687f;   // 1/(sqrt(2)*sqrt(16))

  __syncthreads();

  for (int s0 = 0; s0 < Sc; s0 += CHc) {
    // ---- stage q,k (f32) and V_ext^T (f16) -----------------------------
    for (int i = tid; i < CHc * Fc; i += 512) {
      const int r = i >> 4, c = i & 15;
      const int g = (b * Sc + s0 + r) * (Hc * Fc) + h * Fc + c;
      sm_q[i] = Qg[g];
      sm_k[i] = Kg[g];
    }
    for (int i = tid; i < NEc * CHc; i += 512) {   // i = c*64 + r
      const int c = i >> 6, r = i & 63;
      float vv;
      if (c < DHc)       vv = Vg[(b * Sc + s0 + r) * (Hc * DHc) + h * DHc + c];
      else if (c == DHc) vv = 1.0f;   // ones column -> denominator machinery
      else               vv = 0.0f;
      v_extT[i] = (_Float16)vv;
    }
    __syncthreads();

    // ---- build Taylor feature maps phi(q), phi(k), phi(k)^T ------------
    for (int i = tid; i < CHc * DPc; i += 512) {
      const int r = i / DPc, c = i - r * DPc;
      const float* xq = &sm_q[r * Fc];
      const float* xk = &sm_k[r * Fc];
      float vq, vk;
      if (c == 0)        { vq = 1.0f; vk = 1.0f; }
      else if (c <= Fc)  { vq = xq[c - 1] * inv_xn; vk = xk[c - 1] * inv_xn; }
      else if (c < 273)  {
        const int idx = c - 17, ii = idx >> 4, jj = idx & 15;
        vq = xq[ii] * xq[jj] * inv_x2n;
        vk = xk[ii] * xk[jj] * inv_x2n;
      } else             { vq = 0.0f; vk = 0.0f; }   // padding 273..287
      phi_q[i] = (_Float16)vq;
      phi_k[i] = (_Float16)vk;
      phi_kT[c * CHc + r] = (_Float16)vk;
    }
    __syncthreads();

    // ---- GEMM1: A = tril(phi_q @ phi_k^T), 4x4 tiles, one per wave -----
    {
      const int ti = wave >> 2, tj = wave & 3;
      v8f acc = {};
      if (ti >= tj) {                // upper-triangular tiles are all zero
        for (int k = 0; k < DPc; k += 32) {
          const v16h a  = load_frag_vec(phi_q, ti * 16, k, DPc, lane);
          const v16h bf = load_frag_vec(phi_k, tj * 16, k, DPc, lane); // phi_k^T
          acc = wmma_f32_f16(a, bf, acc);
        }
      }
      const int n = lane & 15, hi = (lane >> 4) << 3;
#pragma unroll
      for (int r = 0; r < 8; ++r) {
        const int mrow = r + hi;
        float v = acc[r];
        if (ti == tj && mrow < n) v = 0.0f;   // causal mask (keep diagonal)
        a_m[(ti * 16 + mrow) * CHc + tj * 16 + n] = (_Float16)v;
      }
    }
    __syncthreads();

    // ---- GEMM2 (fused): O_ext = A@V_ext + phi_q@State_ext, 4x5 tiles ---
    v8f oacc[2] = {{}, {}};
    const int tcnt = (wave < 4) ? 2 : 1;      // waves 0..3 also own denom tiles
    for (int t2 = 0; t2 < tcnt; ++t2) {
      const int t  = wave + 16 * t2;          // 0..19
      const int ti = t & 3, tc = t >> 2;      // tc==4 -> cols 64..79 (denominator)
      v8f acc = {};
      for (int k = 0; k < CHc; k += 32) {     // intra-chunk part
        const v16h a  = load_frag_vec(a_m, ti * 16, k, CHc, lane);
        const v16h bf = load_frag_vec(v_extT, tc * 16, k, CHc, lane);
        acc = wmma_f32_f16(a, bf, acc);
      }
      for (int k = 0; k < DPc; k += 32) {     // inter-chunk part via state
        const v16h a  = load_frag_vec(phi_q, ti * 16, k, DPc, lane);
        const v16h bf = load_frag_vec(st16T, tc * 16, k, DPc, lane);
        acc = wmma_f32_f16(a, bf, acc);
      }
      oacc[t2] = acc;
      if (tc == 4 && (lane & 15) == 0) {      // extract denominator (local col 0)
        const int hi = (lane >> 4) << 3;
#pragma unroll
        for (int r = 0; r < 8; ++r) sm_den[ti * 16 + r + hi] = acc[r];
      }
    }
    __syncthreads();

    // ---- normalize and write per-head output (f16) ---------------------
    for (int t2 = 0; t2 < tcnt; ++t2) {
      const int t  = wave + 16 * t2;
      const int ti = t & 3, tc = t >> 2;
      if (tc >= 4) continue;
      const int n = lane & 15, hi = (lane >> 4) << 3;
#pragma unroll
      for (int r = 0; r < 8; ++r) {
        const int row = ti * 16 + r + hi;
        const float val = oacc[t2][r] / (sm_den[row] + 1e-9f);
        Og[(b * Sc + s0 + row) * (Hc * DHc) + h * DHc + tc * 16 + n] =
            (_Float16)val;
      }
    }

    // ---- GEMM3: State_ext += phi_k^T @ V_ext (18x5 tiles over 16 waves) -
    // st16T reads for this chunk finished before the last barrier.
#pragma unroll
    for (int j = 0; j < 6; ++j) {
      const int t = wave + 16 * j;
      if (t >= 90) break;
      const int tm = t % 18, tc = t / 18;
      v8f acc = stacc[j];
      for (int k = 0; k < CHc; k += 32) {
        const v16h a  = load_frag_vec(phi_kT, tm * 16, k, CHc, lane); // phi_k^T
        const v16h bf = load_frag_vec(v_extT, tc * 16, k, CHc, lane);
        acc = wmma_f32_f16(a, bf, acc);
      }
      stacc[j] = acc;
      // transposed mirror write: rows of D are contiguous -> one B128 store
      const int n = lane & 15, hi = (lane >> 4) << 3;
      v8h sv;
#pragma unroll
      for (int r = 0; r < 8; ++r) sv[r] = (_Float16)acc[r];
      *(v8h*)(st16T + (tc * 16 + n) * DPc + tm * 16 + hi) = sv;
    }
    __syncthreads();   // protect phi/v_extT/st16T before next chunk's staging
  }
}

// ---------------------------------------------------------------------------
// Host-side orchestration
// ---------------------------------------------------------------------------
static constexpr size_t OFF_HS16 = 0;
static constexpr size_t OFF_WQ16 = OFF_HS16 + (size_t)Bc * Sc * DMc * 2;
static constexpr size_t OFF_WK16 = OFF_WQ16 + (size_t)DMc * (Hc * Fc) * 2;
static constexpr size_t OFF_WV16 = OFF_WK16 + (size_t)DMc * (Hc * Fc) * 2;
static constexpr size_t OFF_WO16 = OFF_WV16 + (size_t)DMc * DMc * 2;
static constexpr size_t OFF_QF   = OFF_WO16 + (size_t)DMc * DMc * 2;
static constexpr size_t OFF_KF   = OFF_QF + (size_t)Bc * Sc * (Hc * Fc) * 4;
static constexpr size_t OFF_VF   = OFF_KF + (size_t)Bc * Sc * (Hc * Fc) * 4;
static constexpr size_t OFF_O16  = OFF_VF + (size_t)Bc * Sc * DMc * 4;

static constexpr int ATTN_SMEM_BYTES =
    (2 * CHc * Fc + CHc) * 4 +
    (2 * CHc * DPc + DPc * CHc + NEc * CHc + CHc * CHc + NEc * DPc) * 2; // 183552

extern "C" void kernel_launch(void* const* d_in, const int* in_sizes, int n_in,
                              void* d_out, int out_size, void* d_ws, size_t ws_size,
                              hipStream_t stream) {
  (void)in_sizes; (void)n_in; (void)out_size; (void)ws_size;

  const float* hs = (const float*)d_in[0];
  const float* Wq = (const float*)d_in[1];
  const float* Wk = (const float*)d_in[2];
  const float* Wv = (const float*)d_in[3];
  const float* Wo = (const float*)d_in[4];
  float* out = (float*)d_out;

  char* ws = (char*)d_ws;
  _Float16* hs16  = (_Float16*)(ws + OFF_HS16);
  _Float16* wq16T = (_Float16*)(ws + OFF_WQ16);   // [N,K] transposed
  _Float16* wk16T = (_Float16*)(ws + OFF_WK16);
  _Float16* wv16T = (_Float16*)(ws + OFF_WV16);
  _Float16* wo16T = (_Float16*)(ws + OFF_WO16);
  float*    Qf    = (float*)(ws + OFF_QF);
  float*    Kf    = (float*)(ws + OFF_KF);
  float*    Vf    = (float*)(ws + OFF_VF);
  _Float16* O16   = (_Float16*)(ws + OFF_O16);

  const int M = Bc * Sc;          // 4096 rows
  // conversions: activations row-major, weights transposed to [N,K]
  cvt_f32_to_f16_kernel<<<512, 256, 0, stream>>>(hs, hs16, M * DMc);
  cvt_transpose_f16_kernel<<<256, 256, 0, stream>>>(Wq, wq16T, DMc, Hc * Fc);
  cvt_transpose_f16_kernel<<<256, 256, 0, stream>>>(Wk, wk16T, DMc, Hc * Fc);
  cvt_transpose_f16_kernel<<<512, 256, 0, stream>>>(Wv, wv16T, DMc, DMc);
  cvt_transpose_f16_kernel<<<512, 256, 0, stream>>>(Wo, wo16T, DMc, DMc);

  // projections: Q/K [4096,256], V [4096,1024]
  {
    const int blocksQK = ((M / 16) * ((Hc * Fc) / 64)) / 8;   // 128
    const int blocksV  = ((M / 16) * (DMc / 64)) / 8;         // 512
    wmma_gemm_kernel<<<blocksQK, 256, 0, stream>>>(hs16, wq16T, Qf, M, Hc * Fc, DMc);
    wmma_gemm_kernel<<<blocksQK, 256, 0, stream>>>(hs16, wk16T, Kf, M, Hc * Fc, DMc);
    wmma_gemm_kernel<<<blocksV,  256, 0, stream>>>(hs16, wv16T, Vf, M, DMc, DMc);
  }

  // chunked causal linear attention: one workgroup per (b,h)
  (void)hipFuncSetAttribute((const void*)linattn_kernel,
                            hipFuncAttributeMaxDynamicSharedMemorySize,
                            ATTN_SMEM_BYTES);
  linattn_kernel<<<Bc * Hc, 512, ATTN_SMEM_BYTES, stream>>>(Qf, Kf, Vf, O16);

  // output projection: out = O16 @ Wo  -> [4096,1024] f32
  {
    const int blocksO = ((M / 16) * (DMc / 64)) / 8;          // 512
    wmma_gemm_kernel<<<blocksO, 256, 0, stream>>>(O16, wo16T, out, M, DMc, DMc);
  }
}
